// AngularFeaturePropagation_1846835937529
// MI455X (gfx1250) — matrix operand order repaired and checked by reading, not compiled, not run on hardware
//
#include <hip/hip_runtime.h>
#include <hip/hip_bf16.h>

// ---------------------------------------------------------------------------
// AngularFeaturePropagation for MI455X (gfx1250, wave32, WMMA)
// B=8, N=8192, M=2048, C1=64, C2=128, MLP=[128,128]
// ---------------------------------------------------------------------------

#define BN_ 8
#define NN_ 8192
#define MM_ 2048
#define C1_ 64
#define C2_ 128
#define CIN_ 192      // C1 + C2
#define COUT_ 128
#define EPS_ 1e-5f

typedef _Float16 v16h __attribute__((ext_vector_type(16)));
typedef float    v8f  __attribute__((ext_vector_type(8)));

union Frag16 { uint4 u[2]; v16h h; };   // 32 bytes = one 16-bit WMMA operand per lane

// ---------------------------------------------------------------------------
// 0) zero the stats region (atomics accumulate; must clear every call)
// ---------------------------------------------------------------------------
__global__ void zero_stats_kernel(float* s) {
    s[threadIdx.x] = 0.0f;   // 512 threads: sum0,sumsq0,sum1,sumsq1
}

// ---------------------------------------------------------------------------
// 1) convert weights f32 -> f16 (tiny, stays in L2)
// ---------------------------------------------------------------------------
__global__ void cvt_weights_kernel(const float* __restrict__ W0,
                                   const float* __restrict__ W1,
                                   _Float16* __restrict__ W0h,
                                   _Float16* __restrict__ W1h) {
    int i = blockIdx.x * 256 + threadIdx.x;
    if (i < COUT_ * CIN_)  W0h[i] = (_Float16)W0[i];
    if (i < COUT_ * COUT_) W1h[i] = (_Float16)W1[i];
}

// ---------------------------------------------------------------------------
// 2) fused NN-argmin + gather + concat -> cat_f16 [b][192][N] (n contiguous)
//    argmin of squared distance == argmin of sqrt distance; strict '<'
//    keeps the first minimum (jnp.argmin tie-break).
// ---------------------------------------------------------------------------
__global__ void __launch_bounds__(256)
nn_gather_cat_kernel(const float* __restrict__ lt, const float* __restrict__ lp,
                     const float* __restrict__ lf,
                     const float* __restrict__ ht, const float* __restrict__ hp,
                     const float* __restrict__ hf,
                     _Float16* __restrict__ cat) {
    __shared__ float st[MM_];
    __shared__ float sp[MM_];
    const int b = blockIdx.y;
    const int n = blockIdx.x * 256 + threadIdx.x;

    for (int m = threadIdx.x; m < MM_; m += 256) {
        st[m] = ht[b * MM_ + m];
        sp[m] = hp[b * MM_ + m];
    }
    __syncthreads();

    const float t = lt[b * NN_ + n];
    const float p = lp[b * NN_ + n];
    float best = 3.4e38f;
    int bidx = 0;
#pragma unroll 4
    for (int m = 0; m < MM_; ++m) {
        const float dt = t - st[m];
        const float dp = p - sp[m];
        const float d = fmaf(dt, dt, dp * dp);
        if (d < best) { best = d; bidx = m; }
    }

    _Float16* cc = cat + (size_t)b * CIN_ * NN_ + n;
    const float* lfb = lf + (size_t)b * C1_ * NN_ + n;
#pragma unroll 4
    for (int i = 0; i < C1_; ++i)
        cc[(size_t)i * NN_] = (_Float16)lfb[(size_t)i * NN_];

    const float* hfb = hf + (size_t)b * C2_ * MM_ + bidx;
    _Float16* ch = cc + (size_t)C1_ * NN_;
#pragma unroll 4
    for (int c = 0; c < C2_; ++c)
        ch[(size_t)c * NN_] = (_Float16)hfb[(size_t)c * MM_];
}

// ---------------------------------------------------------------------------
// 3) WMMA GEMM + bias + BN-stat accumulation.
//    Y[b][o][n] = sum_k W[o][k] * X[b][k][n] + bias[o]
//    Block: 256 threads (8 waves), one batch, one 128-column strip.
//    Wave w owns output channels [16w, 16w+16); loops 8 n-tiles of 16.
//    B operand staged through LDS as [n][k] (padded stride, 16B aligned)
//    so each fragment is two ds_load_b128; A comes from the f16 weights
//    in L2 as two global_load_b128 per k-step (contiguous k order, 16 k per lane half).
// ---------------------------------------------------------------------------
template <int K>
__global__ void __launch_bounds__(256)
gemm_bn_kernel(const _Float16* __restrict__ W,   // [128][K] row-major, f16
               const float* __restrict__ bias,   // [128]
               const _Float16* __restrict__ X,   // [B][K][N], f16
               float* __restrict__ Y,            // [B][128][N], f32
               float* __restrict__ sum, float* __restrict__ sumsq) {
    constexpr int KP = K + 8;                    // pad: KP*2 bytes multiple of 16
    constexpr int KSTEPS = K / 32;
    __shared__ __align__(16) _Float16 sh[128 * KP];

    const int b  = blockIdx.y;
    const int n0 = blockIdx.x * 128;
    const _Float16* Xb = X + (size_t)b * K * NN_;

    // stage X tile [K][128] -> LDS transposed [n][k]
    for (int e = threadIdx.x; e < K * 64; e += 256) {
        const int k  = e >> 6;
        const int nn = (e & 63) * 2;
        union { unsigned int u; _Float16 h[2]; } v;
        v.u = *reinterpret_cast<const unsigned int*>(Xb + (size_t)k * NN_ + n0 + nn);
        sh[nn * KP + k]       = v.h[0];
        sh[(nn + 1) * KP + k] = v.h[1];
    }
    __syncthreads();

    const int lane = threadIdx.x & 31;
    const int wave = threadIdx.x >> 5;
    const int o0   = wave * 16;
    const int m    = lane & 15;
    const int hi   = (lane >> 4) & 1;

    // A fragments for all k-steps, held in registers across the n-tile loop.
    // A k order used here (contiguous): lanes 0-15 hold K {0..15}, lanes 16-31 {16..31}.
    Frag16 afrag[KSTEPS];
    const _Float16* Wrow = W + (o0 + m) * K;
#pragma unroll
    for (int ks = 0; ks < KSTEPS; ++ks) {
        const int kb = ks * 32 + hi * 16;
        const uint4* pa = reinterpret_cast<const uint4*>(Wrow + kb);
        afrag[ks].u[0] = pa[0];   // K = kb .. kb+7
        afrag[ks].u[1] = pa[1];   // K = kb+8 .. kb+15
    }

    float bv[8];
#pragma unroll
    for (int r = 0; r < 8; ++r) bv[r] = bias[o0 + hi * 8 + r];

    float s1[8] = {0, 0, 0, 0, 0, 0, 0, 0};
    float s2[8] = {0, 0, 0, 0, 0, 0, 0, 0};

    for (int nt = 0; nt < 8; ++nt) {
        v8f acc = {0.f, 0.f, 0.f, 0.f, 0.f, 0.f, 0.f, 0.f};
#pragma unroll
        for (int ks = 0; ks < KSTEPS; ++ks) {
            // B k order used here (same as A): lane holds column (lane&15), K = hi*16 + e
            Frag16 bfrag;
            const int kb = ks * 32 + hi * 16;
            const uint4* pb =
                reinterpret_cast<const uint4*>(&sh[(nt * 16 + m) * KP + kb]);
            bfrag.u[0] = pb[0];
            bfrag.u[1] = pb[1];
            acc = __builtin_amdgcn_wmma_f32_16x16x32_f16(
                false, afrag[ks].h, false, bfrag.h, (short)0, acc, false, false);
        }
        const int n = n0 + nt * 16 + m;
        float* Yp = Y + ((size_t)b * COUT_ + o0 + hi * 8) * NN_ + n;
#pragma unroll
        for (int r = 0; r < 8; ++r) {        // D layout: VGPR r -> row r (+8 for hi half)
            const float v = acc[r] + bv[r];
            Yp[(size_t)r * NN_] = v;
            s1[r] += v;
            s2[r] += v * v;
        }
    }

    // reduce over the 16 lanes of each half-wave, then 2 atomics per r
#pragma unroll
    for (int r = 0; r < 8; ++r) {
        float a1 = s1[r], a2 = s2[r];
        for (int off = 1; off < 16; off <<= 1) {
            a1 += __shfl_xor(a1, off, 32);
            a2 += __shfl_xor(a2, off, 32);
        }
        if ((lane & 15) == 0) {
            const int o = o0 + hi * 8 + r;
            atomicAdd(&sum[o], a1);
            atomicAdd(&sumsq[o], a2);
        }
    }
}

// ---------------------------------------------------------------------------
// 4) per-channel BN finalize: scale = g*rsqrt(var+eps), shift = beta - mean*scale
// ---------------------------------------------------------------------------
__global__ void bn_finalize_kernel(const float* __restrict__ sum,
                                   const float* __restrict__ sumsq,
                                   const float* __restrict__ g,
                                   const float* __restrict__ beta,
                                   float* __restrict__ scale,
                                   float* __restrict__ shift) {
    const int o = threadIdx.x;              // 128 threads
    const float cnt = (float)(BN_ * NN_);
    const float mean = sum[o] / cnt;
    const float var  = fmaxf(sumsq[o] / cnt - mean * mean, 0.0f);
    const float inv  = rsqrtf(var + EPS_);
    const float sc   = g[o] * inv;
    scale[o] = sc;
    shift[o] = beta[o] - mean * sc;
}

// ---------------------------------------------------------------------------
// 5a) normalize + ReLU, emit f16 activations for the next GEMM
// ---------------------------------------------------------------------------
__global__ void __launch_bounds__(256)
norm_relu_f16_kernel(const float* __restrict__ Y,
                     const float* __restrict__ scale,
                     const float* __restrict__ shift,
                     _Float16* __restrict__ H) {
    const size_t idx = (size_t)blockIdx.x * 256 + threadIdx.x;
    const int o = (int)((idx >> 13) & 127);   // N = 8192 = 2^13
    const float v = fmaxf(fmaf(Y[idx], scale[o], shift[o]), 0.0f);
    H[idx] = (_Float16)v;
}

// 5b) final normalize + ReLU, f32 in place on d_out
__global__ void __launch_bounds__(256)
norm_relu_inplace_kernel(float* __restrict__ Y,
                         const float* __restrict__ scale,
                         const float* __restrict__ shift) {
    const size_t idx = (size_t)blockIdx.x * 256 + threadIdx.x;
    const int o = (int)((idx >> 13) & 127);
    Y[idx] = fmaxf(fmaf(Y[idx], scale[o], shift[o]), 0.0f);
}

// ---------------------------------------------------------------------------
extern "C" void kernel_launch(void* const* d_in, const int* in_sizes, int n_in,
                              void* d_out, int out_size, void* d_ws, size_t ws_size,
                              hipStream_t stream) {
    const float* low_theta  = (const float*)d_in[0];
    const float* low_phi    = (const float*)d_in[1];
    const float* low_feats  = (const float*)d_in[2];
    const float* high_theta = (const float*)d_in[3];
    const float* high_phi   = (const float*)d_in[4];
    const float* high_feats = (const float*)d_in[5];
    const float* W0    = (const float*)d_in[6];
    const float* b0    = (const float*)d_in[7];
    const float* g0    = (const float*)d_in[8];
    const float* beta0 = (const float*)d_in[9];
    const float* W1    = (const float*)d_in[10];
    const float* b1    = (const float*)d_in[11];
    const float* g1    = (const float*)d_in[12];
    const float* beta1 = (const float*)d_in[13];
    float* out = (float*)d_out;               // [B][128][N] f32; also GEMM staging

    // workspace layout
    char* ws = (char*)d_ws;
    float* sum0   = (float*)ws;               // [128]
    float* sumsq0 = sum0 + 128;
    float* sum1   = sum0 + 256;
    float* sumsq1 = sum0 + 384;
    float* scale0 = sum0 + 512;
    float* shift0 = sum0 + 640;
    float* scale1 = sum0 + 768;
    float* shift1 = sum0 + 896;
    _Float16* W0h  = (_Float16*)(ws + 4096);                    // 128*192*2 = 49152 B
    _Float16* W1h  = (_Float16*)(ws + 4096 + 49152);            // 128*128*2 = 32768 B
    _Float16* catf = (_Float16*)(ws + 4096 + 49152 + 32768);    // [B][192][N] f16, 25.2 MB
    _Float16* hbuf = (_Float16*)(ws + 4096 + 49152 + 32768 +
                                 (size_t)BN_ * CIN_ * NN_ * 2); // [B][128][N] f16, 16.8 MB

    zero_stats_kernel<<<1, 512, 0, stream>>>(sum0);
    cvt_weights_kernel<<<(COUT_ * CIN_ + 255) / 256, 256, 0, stream>>>(W0, W1, W0h, W1h);

    nn_gather_cat_kernel<<<dim3(NN_ / 256, BN_), 256, 0, stream>>>(
        low_theta, low_phi, low_feats, high_theta, high_phi, high_feats, catf);

    // layer 0: y0 -> d_out (staging), stats -> sum0/sumsq0
    gemm_bn_kernel<CIN_><<<dim3(NN_ / 128, BN_), 256, 0, stream>>>(
        W0h, b0, catf, out, sum0, sumsq0);
    bn_finalize_kernel<<<1, 128, 0, stream>>>(sum0, sumsq0, g0, beta0, scale0, shift0);
    norm_relu_f16_kernel<<<(BN_ * COUT_ * NN_) / 256, 256, 0, stream>>>(
        out, scale0, shift0, hbuf);

    // layer 1: y1 overwrites d_out, then normalized in place
    gemm_bn_kernel<COUT_><<<dim3(NN_ / 128, BN_), 256, 0, stream>>>(
        W1h, b1, hbuf, out, sum1, sumsq1);
    bn_finalize_kernel<<<1, 128, 0, stream>>>(sum1, sumsq1, g1, beta1, scale1, shift1);
    norm_relu_inplace_kernel<<<(BN_ * COUT_ * NN_) / 256, 256, 0, stream>>>(
        out, scale1, shift1);
}
